// FastQuantumCircuit_45904610460081
// MI455X (gfx1250) — matrix operand order repaired and checked by reading, not compiled, not run on hardware
//
#include <hip/hip_runtime.h>

// ---------------------------------------------------------------------------
// FastQuantumCircuit on MI455X (gfx1250):
//   Pass 1: simulate the 1024 basis states through the (sample-independent)
//           variational circuit -> real-representation matrix Bt (f16, 8MB) in ws.
//   Pass 2: per-16-sample workgroup: build product-state encoding rows in LDS,
//           GEMM 16x2048x2048 with v_wmma_f32_16x16x32_f16 (two independent
//           accumulator tiles per wave), reduce |amp|^2 -> <Z_i>.
// ---------------------------------------------------------------------------

#define NQ      10
#define NSTATE  1024
#define RDIM    2048          // real-representation dimension (Re/Im interleaved)

typedef _Float16 v16h __attribute__((ext_vector_type(16)));
typedef _Float16 v8h  __attribute__((ext_vector_type(8)));
typedef _Float16 v2h  __attribute__((ext_vector_type(2)));
typedef float    v8f  __attribute__((ext_vector_type(8)));

__device__ __forceinline__ float2 cmul(float2 a, float2 b) {
  return make_float2(a.x * b.x - a.y * b.y, a.x * b.y + a.y * b.x);
}
__device__ __forceinline__ float2 cadd(float2 a, float2 b) {
  return make_float2(a.x + b.x, a.y + b.y);
}

template <int I> struct Int { static constexpr int v = I; };
template <int I, int N, typename F>
__device__ __forceinline__ void static_for(F&& f) {
  if constexpr (I < N) { f(Int<I>{}); static_for<I + 1, N>((F&&)f); }
}

// ---------------------------------------------------------------------------
// Register-resident 10-qubit statevector per wave32:
//   amplitude index j (bit 9-i <-> wire i, matching the reference flatten order)
//   lane = j[9:5], register r = j[4:0]  ->  float2 s[32] = 64 VGPRs.
// All indices compile-time (templates + full unroll) => zero scratch.
// ---------------------------------------------------------------------------
template <int TB>
__device__ __forceinline__ void apply_1q(float2 (&s)[32], int lane,
                                         float2 u00, float2 u01,
                                         float2 u10, float2 u11) {
  if constexpr (TB < 5) {                       // pair lives inside the lane
    constexpr int m = 1 << TB;
#pragma unroll
    for (int r = 0; r < 32; ++r) {
      if ((r & m) == 0) {
        float2 a0 = s[r], a1 = s[r | m];
        s[r]     = cadd(cmul(u00, a0), cmul(u01, a1));
        s[r | m] = cadd(cmul(u10, a0), cmul(u11, a1));
      }
    }
  } else {                                      // pair spans lanes: shfl_xor
    constexpr int lm = 1 << (TB - 5);
    const bool hi = (lane & lm) != 0;
#pragma unroll
    for (int r = 0; r < 32; ++r) {
      float2 o;
      o.x = __shfl_xor(s[r].x, lm, 32);
      o.y = __shfl_xor(s[r].y, lm, 32);
      float2 lo_a = hi ? o : s[r];
      float2 hi_a = hi ? s[r] : o;
      float2 r0 = cadd(cmul(u00, lo_a), cmul(u01, hi_a));
      float2 r1 = cadd(cmul(u10, lo_a), cmul(u11, hi_a));
      s[r] = hi ? r1 : r0;
    }
  }
}

template <int CB, int TB>
__device__ __forceinline__ void apply_cnot(float2 (&s)[32], int lane) {
  if constexpr (TB < 5) {
    constexpr int tm = 1 << TB;
    if constexpr (CB < 5) {
#pragma unroll
      for (int r = 0; r < 32; ++r) {
        if (((r & tm) == 0) && ((r & (1 << CB)) != 0)) {
          float2 tmp = s[r]; s[r] = s[r | tm]; s[r | tm] = tmp;
        }
      }
    } else {
      const bool ctrl = (lane & (1 << (CB - 5))) != 0;
#pragma unroll
      for (int r = 0; r < 32; ++r) {
        if ((r & tm) == 0) {
          float2 t0 = s[r], t1 = s[r | tm];
          s[r]      = ctrl ? t1 : t0;
          s[r | tm] = ctrl ? t0 : t1;
        }
      }
    }
  } else {
    constexpr int lm = 1 << (TB - 5);
#pragma unroll
    for (int r = 0; r < 32; ++r) {
      float2 o;
      o.x = __shfl_xor(s[r].x, lm, 32);
      o.y = __shfl_xor(s[r].y, lm, 32);
      bool ctrl;
      if constexpr (CB < 5) ctrl = (r & (1 << CB)) != 0;
      else                  ctrl = (lane & (1 << (CB - 5))) != 0;
      s[r] = ctrl ? o : s[r];
    }
  }
}

// ---------------------------------------------------------------------------
// Pass 1: one wave per basis state k. Output Bt (2048 x 2048 f16, row-major),
// Bt[c][r] = real-rep matrix B^T such that W[s,c] = sum_r V[s,r]*Bt[c][r]:
//   Bt[2j  ][2k] =  Re U[j,k];  Bt[2j  ][2k+1] = -Im U[j,k]
//   Bt[2j+1][2k] =  Im U[j,k];  Bt[2j+1][2k+1] =  Re U[j,k]
// ---------------------------------------------------------------------------
__launch_bounds__(256)
__global__ void qc_build_unitary(const float* __restrict__ wt,
                                 _Float16* __restrict__ Bt) {
  const int k    = blockIdx.x * 8 + (threadIdx.x >> 5);   // basis state 0..1023
  const int lane = threadIdx.x & 31;
  const int khi = k >> 5, klo = k & 31;

  float2 s[32];
#pragma unroll
  for (int r = 0; r < 32; ++r) {
    s[r].x = (lane == khi && r == klo) ? 1.f : 0.f;
    s[r].y = 0.f;
  }

  static_for<0, 4>([&](auto L) {
    constexpr int layer = decltype(L)::v;
    static_for<0, NQ>([&](auto W) {
      constexpr int w = decltype(W)::v;
      const float a  = wt[(layer * NQ + w) * 3 + 0];
      const float b  = wt[(layer * NQ + w) * 3 + 1];
      const float cc = wt[(layer * NQ + w) * 3 + 2];
      const float cb = cosf(0.5f * b), sb = sinf(0.5f * b);
      const float apc = 0.5f * (a + cc), amc = 0.5f * (a - cc);
      const float cp = cosf(apc), sp = sinf(apc);
      const float cm = cosf(amc), sm = sinf(amc);
      // U = Rz(cc)*Ry(b)*Rz(a)
      const float2 u00 = make_float2( cb * cp, -cb * sp);
      const float2 u01 = make_float2(-sb * cm, -sb * sm);
      const float2 u10 = make_float2( sb * cm, -sb * sm);
      const float2 u11 = make_float2( cb * cp,  cb * sp);
      apply_1q<9 - w>(s, lane, u00, u01, u10, u11);
    });
    constexpr int off = (layer % 2 == 0) ? 1 : (NQ / 2);
    static_for<0, NQ>([&](auto W) {
      constexpr int w = decltype(W)::v;
      apply_cnot<9 - w, 9 - ((w + off) % NQ)>(s, lane);
    });
  });

#pragma unroll
  for (int r = 0; r < 32; ++r) {
    const int j = (lane << 5) | r;          // output amplitude index
    const size_t b0 = (size_t)(2 * j) * RDIM + 2 * k;
    v2h lo = { (_Float16)s[r].x, (_Float16)(-s[r].y) };
    v2h hi = { (_Float16)s[r].y, (_Float16)( s[r].x) };
    *(v2h*)(Bt + b0)        = lo;
    *(v2h*)(Bt + b0 + RDIM) = hi;
  }
}

// ---------------------------------------------------------------------------
// Pass 2: one workgroup (4 waves / 128 threads) per 16 samples.
//  - build 16 encoding rows (f16, XOR-swizzled 16B blocks) into 64KB LDS
//  - each wave sweeps PAIRS of 16-column strips: shared A fragment, two
//    independent accumulators -> no WMMA->WMMA RAW chain on the matrix pipe
//  - epilogue folds |Re+iIm|^2 into per-lane <Z_i> accumulators via FMA
//    against a per-strip precomputed sign vector
// ---------------------------------------------------------------------------
__launch_bounds__(128)
__global__ void qc_encode_gemm(const float* __restrict__ x,
                               const _Float16* __restrict__ Bt,
                               float* __restrict__ out) {
  __shared__ __align__(16) _Float16 Ash[16 * RDIM];   // 64 KB (of 320KB/WGP)

  const int t  = threadIdx.x;
  const int s0 = blockIdx.x * 16;

  // ---- encoding: product state per sample (8 threads per sample) ----
  {
    const int m = t >> 3, sub = t & 7;
    const float* xs = x + (size_t)(s0 + m) * (2 * NQ);
    float2 f0[NQ], f1[NQ];
    const float PIC = 3.14159265358979f, IS2 = 0.70710678118655f;
#pragma unroll
    for (int q = 0; q < NQ; ++q) {
      const float th = 0.5f * PIC * (xs[q] + 1.f);
      const float ph = 0.5f * PIC * (xs[NQ + q] + 1.f);
      const float ct = cosf(0.5f * th), st = sinf(0.5f * th);
      const float g0 = (ct - st) * IS2, g1 = (ct + st) * IS2;  // Ry*H|0>
      const float cp = cosf(0.5f * ph), sp = sinf(0.5f * ph);
      f0[q] = make_float2(g0 * cp, -g0 * sp);                  // * e^{-i ph/2}
      f1[q] = make_float2(g1 * cp,  g1 * sp);                  // * e^{+i ph/2}
    }
    for (int qq = 0; qq < 128; ++qq) {
      const int j = sub + (qq << 3);                 // amplitude 0..1023
      float2 v = (j & 512) ? f1[0] : f0[0];          // wire 0 <-> bit 9
#pragma unroll
      for (int w = 1; w < NQ; ++w)
        v = cmul(v, ((j >> (9 - w)) & 1) ? f1[w] : f0[w]);
      // XOR-swizzle 16B blocks for conflict-free ds_load_b128 fragment reads
      const int idx = m * RDIM + ((((j >> 2) ^ m) & 0xFFFF) << 3) + ((j & 3) << 1);
      Ash[idx]     = (_Float16)v.x;
      Ash[idx + 1] = (_Float16)v.y;
    }
  }
  __syncthreads();

  // ---- GEMM + <Z> reduction ----
  const int lane = t & 31, wave = t >> 5;
  const int hs = lane >> 4, ln = lane & 15;

  float zacc[8][NQ];
#pragma unroll
  for (int r = 0; r < 8; ++r)
#pragma unroll
    for (int q = 0; q < NQ; ++q) zacc[r][q] = 0.f;

  const _Float16* __restrict__ arow = Ash + ln * RDIM;

  // 64 strip-pairs of (16+16) columns; each wave owns pair indices wave,wave+4,...
  for (int sp = wave; sp < 64; sp += 4) {
    const int colbase = sp << 5;                       // first strip's base col
    v8f c0 = {0.f, 0.f, 0.f, 0.f, 0.f, 0.f, 0.f, 0.f};
    v8f c1 = {0.f, 0.f, 0.f, 0.f, 0.f, 0.f, 0.f, 0.f};
    const _Float16* __restrict__ brow0 =
        Bt + (size_t)(colbase + ln) * RDIM + hs * 8;  // B-frag: row of Bt
    const _Float16* __restrict__ brow1 = brow0 + (size_t)16 * RDIM;

#pragma unroll 4
    for (int kb = 0; kb < 64; ++kb) {
      const int K0 = kb << 5;
      __builtin_prefetch(brow0 + K0 + 256, 0, 3);      // global_prefetch_b8
      __builtin_prefetch(brow1 + K0 + 256, 0, 3);
      // A fragment (16x32 f16), shared by both strips:
      //   lanes0-15 K0..K0+7 / +16..23, lanes16-31 +8..15 / +24..31
      const int b0 = (K0 >> 3) + hs;
      v8h a0 = *(const v8h*)(arow + (((b0    ) ^ ln) << 3));
      v8h a1 = *(const v8h*)(arow + (((b0 + 2) ^ ln) << 3));
      v16h A = __builtin_shufflevector(a0, a1, 0,1,2,3,4,5,6,7,8,9,10,11,12,13,14,15);
      // B fragments (32x16 f16): lane column N=ln along K, contiguous in Bt row
      v8h p0 = *(const v8h*)(brow0 + K0);
      v8h p1 = *(const v8h*)(brow0 + K0 + 16);
      v16h B0 = __builtin_shufflevector(p0, p1, 0,1,2,3,4,5,6,7,8,9,10,11,12,13,14,15);
      v8h q0 = *(const v8h*)(brow1 + K0);
      v8h q1 = *(const v8h*)(brow1 + K0 + 16);
      v16h B1 = __builtin_shufflevector(q0, q1, 0,1,2,3,4,5,6,7,8,9,10,11,12,13,14,15);
      // two independent accumulator chains keep the matrix pipe busy
      c0 = __builtin_amdgcn_wmma_f32_16x16x32_f16(false, A, false, B0,
                                                  (short)0, c0, false, false);
      c1 = __builtin_amdgcn_wmma_f32_16x16x32_f16(false, A, false, B1,
                                                  (short)0, c1, false, false);
    }

    // epilogue: cols (2t,2t+1) = (Re,Im) of one amplitude; pair-sum via shfl,
    // then FMA against per-strip sign vectors (hoisted out of the r-loop).
    const int jamp0 = (colbase + (ln & 0x0E)) >> 1;
    const int jamp1 = jamp0 + 8;
    float sg0[NQ], sg1[NQ];
#pragma unroll
    for (int qi = 0; qi < NQ; ++qi) {
      sg0[qi] = ((jamp0 >> (9 - qi)) & 1) ? -1.f : 1.f;
      sg1[qi] = ((jamp1 >> (9 - qi)) & 1) ? -1.f : 1.f;
    }
#pragma unroll
    for (int r = 0; r < 8; ++r) {
      const float v0 = c0[r], v1 = c1[r];
      const float e0 = v0 * v0, e1 = v1 * v1;
      const float p0 = 0.5f * (e0 + __shfl_xor(e0, 1, 32));   // |amp|^2 (half/lane)
      const float p1 = 0.5f * (e1 + __shfl_xor(e1, 1, 32));
#pragma unroll
      for (int qi = 0; qi < NQ; ++qi)
        zacc[r][qi] = fmaf(sg1[qi], p1, fmaf(sg0[qi], p0, zacc[r][qi]));
    }
  }

  // ---- reduce to out: reuse Ash as the z scratch (A panel is dead now) ----
  __syncthreads();
  float* zf = (float*)Ash;                 // 16 samples x 10 qubits
  for (int i = t; i < 16 * NQ; i += 128) zf[i] = 0.f;
  __syncthreads();
#pragma unroll
  for (int r = 0; r < 8; ++r) {
    const int mrow = r + (hs << 3);        // C/D layout: M = r + 8*(lane/16)
#pragma unroll
    for (int qi = 0; qi < NQ; ++qi)
      atomicAdd(&zf[mrow * NQ + qi], zacc[r][qi]);
  }
  __syncthreads();
  for (int i = t; i < 16 * NQ; i += 128)
    out[(size_t)s0 * NQ + i] = zf[i];
}

// ---------------------------------------------------------------------------
extern "C" void kernel_launch(void* const* d_in, const int* in_sizes, int n_in,
                              void* d_out, int out_size, void* d_ws, size_t ws_size,
                              hipStream_t stream) {
  (void)in_sizes; (void)n_in; (void)out_size;
  const float* x  = (const float*)d_in[0];   // (16,256,20) f32
  const float* wt = (const float*)d_in[1];   // (4,10,3)    f32
  _Float16* Bt = (_Float16*)d_ws;            // needs 2048*2048*2 = 8 MB scratch
  (void)ws_size;

  // Pass 1: 1024 basis-state simulations (one wave32 each) -> Bt
  qc_build_unitary<<<128, 256, 0, stream>>>(wt, Bt);
  // Pass 2: 256 workgroups x 16 samples: encode + WMMA GEMM + <Z> reduce
  qc_encode_gemm<<<256, 128, 0, stream>>>(x, Bt, (float*)d_out);
}
